// Actor_35493609734569
// MI455X (gfx1250) — matrix operand order repaired and checked by reading, not compiled, not run on hardware
//
#include <hip/hip_runtime.h>

#define NN 10000          // nodes
#define EE 160000         // edges
#define DD 128            // feature dim everywhere
#define NP 10016          // keys padded to multiple of 32
#define KT 313            // NP / 32 key tiles
#define SCALE 0.08838834764831845f   // 1/sqrt(128)

typedef __attribute__((ext_vector_type(16))) __bf16 v16bf;
typedef __attribute__((ext_vector_type(8)))  float  v8f;

union Frag16 {            // one 16-bit WMMA operand (8 VGPRs)
  v16bf v;
  unsigned int d[8];
};

__device__ __forceinline__ unsigned short f2bf(float f) {
  union { float f; unsigned int u; } x; x.f = f;
  unsigned int r;
  if ((x.u & 0x7F800000u) == 0x7F800000u) r = x.u;          // Inf/NaN passthrough
  else r = x.u + 0x7FFFu + ((x.u >> 16) & 1u);              // round-to-nearest-even
  return (unsigned short)(r >> 16);
}

// ---------------------------------------------------------------- utilities

__global__ void zero_u32(unsigned int* p, long long n) {
  long long i = (long long)blockIdx.x * blockDim.x + threadIdx.x;
  long long st = (long long)gridDim.x * blockDim.x;
  for (; i < n; i += st) p[i] = 0u;
}

__global__ void cvt_bf(const float* __restrict__ in, unsigned short* __restrict__ out, int n) {
  int i = blockIdx.x * blockDim.x + threadIdx.x;
  if (i < n) out[i] = f2bf(in[i]);
}

// 128x128 fp32 W -> bf16 W^T (WT[n*128+k] = W[k*128+n]) so B-fragments load as dword pairs
__global__ void wtrans_bf(const float* __restrict__ W, unsigned short* __restrict__ WT) {
  int i = blockIdx.x * blockDim.x + threadIdx.x;   // 16384 threads
  int k = i >> 7, n = i & 127;
  WT[n * 128 + k] = f2bf(W[k * 128 + n]);
}

// V bf16 [NN,128] -> VT bf16 [128, NP]
__global__ void vtrans(const unsigned short* __restrict__ V, unsigned short* __restrict__ VT) {
  int i = blockIdx.x * blockDim.x + threadIdx.x;
  if (i >= NN * DD) return;
  int r = i >> 7, c = i & 127;
  VT[c * NP + r] = V[r * DD + c];
}

// segment_sum: agg[dst[e]] += x[src[e]]
__global__ void scatter_add(const float* __restrict__ x, const int* __restrict__ src,
                            const int* __restrict__ dst, float* __restrict__ agg) {
  int i = blockIdx.x * blockDim.x + threadIdx.x;
  if (i >= EE * DD) return;
  int e = i >> 7, f = i & 127;
  atomicAdd(&agg[dst[e] * DD + f], x[src[e] * DD + f]);
}

// combined bias row: out[t] = b[t] + bg[t] + sum_k g[k]*Wg[k*128+t]   (GD=64)
__global__ void gbias(const float* __restrict__ g, const float* __restrict__ Wg,
                      const float* __restrict__ bg, const float* __restrict__ b,
                      float* __restrict__ out) {
  int t = threadIdx.x;
  float s = b[t] + bg[t];
  #pragma unroll
  for (int k = 0; k < 64; ++k) s += g[k] * Wg[k * 128 + t];
  out[t] = s;
}

// ------------------------------------------------- dual-input WMMA GEMM
// out = act( A@W1 + (B?B@W2:0) + bias ), A/B bf16 [M,128], W*T bf16 transposed [128,128].
// One wave (32 threads) per 16-row tile. M must be a multiple of 16 (10000 = 625*16).
__global__ void __launch_bounds__(32, 1)
gemm_wmma(const unsigned short* __restrict__ A,
          const unsigned short* __restrict__ B,
          const unsigned short* __restrict__ W1T,
          const unsigned short* __restrict__ W2T,
          const float* __restrict__ bias,
          float* __restrict__ outF,
          unsigned short* __restrict__ outBF,
          int relu) {
  const int lane = threadIdx.x;
  const int rowbase = blockIdx.x * 16;
  const int mrow = rowbase + (lane & 15);
  const int half8 = (lane >> 4) << 3;   // +8 K-offset for upper half (A layout)

  Frag16 af[4], bf2[4];
  #pragma unroll
  for (int kk = 0; kk < 4; ++kk)
    #pragma unroll
    for (int v = 0; v < 8; ++v) {
      int k = kk * 32 + ((v >> 2) << 4) + ((v & 3) << 1) + half8;
      af[kk].d[v] = *(const unsigned int*)(A + mrow * DD + k);
      if (B) bf2[kk].d[v] = *(const unsigned int*)(B + mrow * DD + k);
    }

  #pragma unroll
  for (int ct = 0; ct < 8; ++ct) {
    const int col = ct * 16 + (lane & 15);
    const float bval = bias[col];
    v8f acc = {};
    #pragma unroll
    for (int kk = 0; kk < 4; ++kk) {
      Frag16 wf;
      #pragma unroll
      for (int v = 0; v < 8; ++v) {
        int kd = ((v >> 2) << 4) + ((v & 3) << 1) + half8; // B-operand K packing
        wf.d[v] = *(const unsigned int*)(W1T + col * 128 + kk * 32 + kd);
      }
      acc = __builtin_amdgcn_wmma_f32_16x16x32_bf16(false, af[kk].v, false, wf.v,
                                                    (short)0, acc, false, false);
    }
    if (B) {
      #pragma unroll
      for (int kk = 0; kk < 4; ++kk) {
        Frag16 wf;
        #pragma unroll
        for (int v = 0; v < 8; ++v) {
          int kd = ((v >> 2) << 4) + ((v & 3) << 1) + half8;
          wf.d[v] = *(const unsigned int*)(W2T + col * 128 + kk * 32 + kd);
        }
        acc = __builtin_amdgcn_wmma_f32_16x16x32_bf16(false, bf2[kk].v, false, wf.v,
                                                      (short)0, acc, false, false);
      }
    }
    #pragma unroll
    for (int v = 0; v < 8; ++v) {
      float x = acc[v] + bval;
      if (relu) x = fmaxf(x, 0.0f);
      int row = rowbase + v + ((lane >> 4) << 3);         // C/D layout row
      if (outF) outF[row * DD + col] = x;
      outBF[row * DD + col] = f2bf(x);
    }
  }
}

// ------------------------------------------------- flash attention + row mean
// One wave per 16-row Q tile (625 blocks of 32 threads). Online softmax over
// all keys in 32-key tiles; accumulates sum_i softmax(Q_i K^T)V into accum[128].
__global__ void __launch_bounds__(32, 1)
flash_attn(const unsigned short* __restrict__ Qb,   // [NP,128] bf16
           const unsigned short* __restrict__ Kb,   // [NP,128] bf16
           const unsigned short* __restrict__ VT,   // [128,NP] bf16
           float* __restrict__ accum) {
  __shared__ unsigned int pbuf[16 * 32 / 2];              // 16x32 bf16 P tile
  unsigned short* pb16 = (unsigned short*)pbuf;

  const int lane = threadIdx.x;
  const int rowbase = blockIdx.x * 16;
  const int mrow = rowbase + (lane & 15);
  const int halfA = (lane >> 4) << 3;                     // A-layout K offset
  const int halfB = (lane >> 4) << 4;                     // unused; B loads use halfA too

  Frag16 qf[4];
  #pragma unroll
  for (int kk = 0; kk < 4; ++kk)
    #pragma unroll
    for (int v = 0; v < 8; ++v) {
      int k = kk * 32 + ((v >> 2) << 4) + ((v & 3) << 1) + halfA;
      qf[kk].d[v] = *(const unsigned int*)(Qb + mrow * DD + k);
    }

  float m_[8], l_[8], fac[8];
  v8f O[8];
  #pragma unroll
  for (int v = 0; v < 8; ++v) { m_[v] = -3.0e38f; l_[v] = 0.0f; }
  #pragma unroll
  for (int ct = 0; ct < 8; ++ct)
    #pragma unroll
    for (int v = 0; v < 8; ++v) O[ct][v] = 0.0f;

  for (int kt = 0; kt < KT; ++kt) {
    const int keybase = kt * 32;
    v8f S0 = {}, S1 = {};
    // S = Q @ K^T for two 16-key column blocks
    {
      const int key0 = keybase + (lane & 15);
      const int key1 = key0 + 16;
      #pragma unroll
      for (int kk = 0; kk < 4; ++kk) {
        Frag16 b0, b1;
        #pragma unroll
        for (int v = 0; v < 8; ++v) {
          int kd = ((v >> 2) << 4) + ((v & 3) << 1) + halfA;
          b0.d[v] = *(const unsigned int*)(Kb + key0 * DD + kk * 32 + kd);
          b1.d[v] = *(const unsigned int*)(Kb + key1 * DD + kk * 32 + kd);
        }
        S0 = __builtin_amdgcn_wmma_f32_16x16x32_bf16(false, qf[kk].v, false, b0.v,
                                                     (short)0, S0, false, false);
        S1 = __builtin_amdgcn_wmma_f32_16x16x32_bf16(false, qf[kk].v, false, b1.v,
                                                     (short)0, S1, false, false);
      }
    }
    // online softmax (per row = per S-VGPR per half); write P tile to LDS (bf16)
    const bool valid0 = (keybase + (lane & 15)) < NN;
    const bool valid1 = (keybase + 16 + (lane & 15)) < NN;
    #pragma unroll
    for (int v = 0; v < 8; ++v) {
      float s0 = valid0 ? S0[v] * SCALE : -3.0e38f;
      float s1 = valid1 ? S1[v] * SCALE : -3.0e38f;
      float lm = fmaxf(s0, s1);
      #pragma unroll
      for (int d = 1; d < 16; d <<= 1) lm = fmaxf(lm, __shfl_xor(lm, d, 32));
      float mn = fmaxf(m_[v], lm);
      float f = __expf(m_[v] - mn);
      float p0 = valid0 ? __expf(s0 - mn) : 0.0f;
      float p1 = valid1 ? __expf(s1 - mn) : 0.0f;
      float rs = p0 + p1;
      #pragma unroll
      for (int d = 1; d < 16; d <<= 1) rs += __shfl_xor(rs, d, 32);
      l_[v] = l_[v] * f + rs;
      m_[v] = mn;
      fac[v] = f;
      int row = v + ((lane >> 4) << 3);
      pb16[row * 32 + (lane & 15)] = f2bf(p0);
      pb16[row * 32 + 16 + (lane & 15)] = f2bf(p1);
    }
    __syncthreads();
    // reload P in A layout
    Frag16 pf;
    #pragma unroll
    for (int v = 0; v < 8; ++v) {
      int k = ((v >> 2) << 4) + ((v & 3) << 1) + halfA;
      pf.d[v] = *(const unsigned int*)(pb16 + (lane & 15) * 32 + k);
    }
    // O = O*fac + P @ V_tile   (8 output column tiles)
    #pragma unroll
    for (int ct = 0; ct < 8; ++ct) {
      const int col = ct * 16 + (lane & 15);
      Frag16 vf;
      #pragma unroll
      for (int v = 0; v < 8; ++v) {
        int kd = ((v >> 2) << 4) + ((v & 3) << 1) + halfA;
        vf.d[v] = *(const unsigned int*)(VT + col * NP + keybase + kd);
      }
      #pragma unroll
      for (int v = 0; v < 8; ++v) O[ct][v] *= fac[v];
      O[ct] = __builtin_amdgcn_wmma_f32_16x16x32_bf16(false, pf.v, false, vf.v,
                                                      (short)0, O[ct], false, false);
    }
    __syncthreads();
  }

  // rows: out_i = O_i / l_i ; accumulate column sums into accum (mean later)
  #pragma unroll
  for (int v = 0; v < 8; ++v) l_[v] = 1.0f / l_[v];
  #pragma unroll
  for (int ct = 0; ct < 8; ++ct) {
    float cs = 0.0f;
    #pragma unroll
    for (int v = 0; v < 8; ++v) cs += O[ct][v] * l_[v];
    cs += __shfl_xor(cs, 16, 32);
    if (lane < 16) atomicAdd(&accum[ct * 16 + lane], cs);
  }
}

// ------------------------------------------------- tiny MLP head + softmax
__global__ void head(const float* __restrict__ accum,
                     const float* __restrict__ Wo,  const float* __restrict__ bo,
                     const float* __restrict__ W1,  const float* __restrict__ b1,
                     const float* __restrict__ W2,  const float* __restrict__ b2,
                     const float* __restrict__ W3,  const float* __restrict__ b3,
                     float* __restrict__ out) {
  __shared__ float sa[128], st[128], x1[64], x2[32], lg[128], red[2];
  const int t = threadIdx.x;
  sa[t] = accum[t] * (1.0f / (float)NN);
  __syncthreads();
  float s = bo[t];
  #pragma unroll 4
  for (int k = 0; k < 128; ++k) s += sa[k] * Wo[k * 128 + t];
  st[t] = s;
  __syncthreads();
  if (t < 64) {
    float v = b1[t];
    #pragma unroll 4
    for (int k = 0; k < 128; ++k) v += st[k] * W1[k * 64 + t];
    x1[t] = fmaxf(v, 0.0f);
  }
  __syncthreads();
  if (t < 32) {
    float v = b2[t];
    #pragma unroll 4
    for (int k = 0; k < 64; ++k) v += x1[k] * W2[k * 32 + t];
    x2[t] = fmaxf(v, 0.0f);
  }
  __syncthreads();
  float v3 = b3[t];
  #pragma unroll 4
  for (int k = 0; k < 32; ++k) v3 += x2[k] * W3[k * 128 + t];
  lg[t] = v3;
  __syncthreads();
  if (t == 0) {
    float m = -3.0e38f;
    for (int k = 0; k < 128; ++k) m = fmaxf(m, lg[k]);
    red[0] = m;
  }
  __syncthreads();
  float e = __expf(lg[t] - red[0]);
  lg[t] = e;
  __syncthreads();
  if (t == 0) {
    float su = 0.0f;
    for (int k = 0; k < 128; ++k) su += lg[k];
    red[1] = su;
  }
  __syncthreads();
  out[t] = lg[t] / red[1];
}

// ---------------------------------------------------------------- launch

extern "C" void kernel_launch(void* const* d_in, const int* in_sizes, int n_in,
                              void* d_out, int out_size, void* d_ws, size_t ws_size,
                              hipStream_t stream) {
  const float* x     = (const float*)d_in[0];
  const float* g     = (const float*)d_in[1];
  const int*   ei    = (const int*)d_in[2];
  const float* W1r   = (const float*)d_in[3];
  const float* W1l   = (const float*)d_in[4];
  const float* b1    = (const float*)d_in[5];
  const float* W2r   = (const float*)d_in[6];
  const float* W2l   = (const float*)d_in[7];
  const float* b2    = (const float*)d_in[8];
  const float* WQ    = (const float*)d_in[9];
  const float* bQ    = (const float*)d_in[10];
  const float* WK    = (const float*)d_in[11];
  const float* bK    = (const float*)d_in[12];
  const float* WV    = (const float*)d_in[13];
  const float* bV    = (const float*)d_in[14];
  const float* WQg   = (const float*)d_in[15];
  const float* bQg   = (const float*)d_in[16];
  const float* WKg   = (const float*)d_in[17];
  const float* bKg   = (const float*)d_in[18];
  const float* WVg   = (const float*)d_in[19];
  const float* bVg   = (const float*)d_in[20];
  const float* Wo    = (const float*)d_in[21];
  const float* bo    = (const float*)d_in[22];
  const float* Wfc1  = (const float*)d_in[23];
  const float* bfc1  = (const float*)d_in[24];
  const float* Wfc2  = (const float*)d_in[25];
  const float* bfc2  = (const float*)d_in[26];
  const float* Wfc3  = (const float*)d_in[27];
  const float* bfc3  = (const float*)d_in[28];
  const int* src = ei;
  const int* dst = ei + EE;
  float* out = (float*)d_out;

  char* base = (char*)d_ws;
  size_t off = 0;
  auto alloc = [&](size_t bytes) -> char* {
    char* p = base + off;
    off += (bytes + 255) & ~(size_t)255;
    return p;
  };
  unsigned short* x_bf   = (unsigned short*)alloc((size_t)NN * DD * 2);
  float*          aggF   = (float*)alloc((size_t)NN * DD * 4);
  unsigned short* agg_bf = (unsigned short*)alloc((size_t)NN * DD * 2);
  float*          hF     = (float*)alloc((size_t)NN * DD * 4);
  unsigned short* h_bf   = (unsigned short*)alloc((size_t)NN * DD * 2);
  unsigned short* ne_bf  = (unsigned short*)alloc((size_t)NN * DD * 2);
  unsigned short* W1rT   = (unsigned short*)alloc(128 * 128 * 2);
  unsigned short* W1lT   = (unsigned short*)alloc(128 * 128 * 2);
  unsigned short* W2rT   = (unsigned short*)alloc(128 * 128 * 2);
  unsigned short* W2lT   = (unsigned short*)alloc(128 * 128 * 2);
  unsigned short* WQT    = (unsigned short*)alloc(128 * 128 * 2);
  unsigned short* WKT    = (unsigned short*)alloc(128 * 128 * 2);
  unsigned short* WVT    = (unsigned short*)alloc(128 * 128 * 2);
  float*          cb     = (float*)alloc(3 * 128 * 4);
  unsigned short* Qb     = (unsigned short*)alloc((size_t)NP * DD * 2);
  unsigned short* Kb     = (unsigned short*)alloc((size_t)NP * DD * 2);
  unsigned short* Vb     = (unsigned short*)alloc((size_t)NP * DD * 2);
  unsigned short* VTb    = (unsigned short*)alloc((size_t)128 * NP * 2);
  float*          accum  = (float*)alloc(128 * 4);

  const int TPB = 256;
  // zero scratch that must start clean (padded QKV buffers, agg, accum)
  zero_u32<<<2048, TPB, 0, stream>>>((unsigned int*)aggF, (long long)NN * DD);
  zero_u32<<<1, 128, 0, stream>>>((unsigned int*)accum, 128);
  zero_u32<<<2048, TPB, 0, stream>>>((unsigned int*)Qb, (long long)NP * DD / 2);
  zero_u32<<<2048, TPB, 0, stream>>>((unsigned int*)Kb, (long long)NP * DD / 2);
  zero_u32<<<2048, TPB, 0, stream>>>((unsigned int*)Vb, (long long)NP * DD / 2);
  zero_u32<<<2048, TPB, 0, stream>>>((unsigned int*)VTb, (long long)128 * NP / 2);

  // converts
  cvt_bf<<<(NN * DD + TPB - 1) / TPB, TPB, 0, stream>>>(x, x_bf, NN * DD);
  wtrans_bf<<<64, TPB, 0, stream>>>(W1r, W1rT);
  wtrans_bf<<<64, TPB, 0, stream>>>(W1l, W1lT);
  wtrans_bf<<<64, TPB, 0, stream>>>(W2r, W2rT);
  wtrans_bf<<<64, TPB, 0, stream>>>(W2l, W2lT);
  wtrans_bf<<<64, TPB, 0, stream>>>(WQ, WQT);
  wtrans_bf<<<64, TPB, 0, stream>>>(WK, WKT);
  wtrans_bf<<<64, TPB, 0, stream>>>(WV, WVT);

  // GraphConv layer 1
  scatter_add<<<(EE * DD + TPB - 1) / TPB, TPB, 0, stream>>>(x, src, dst, aggF);
  cvt_bf<<<(NN * DD + TPB - 1) / TPB, TPB, 0, stream>>>(aggF, agg_bf, NN * DD);
  gemm_wmma<<<NN / 16, 32, 0, stream>>>(x_bf, agg_bf, W1rT, W1lT, b1, hF, h_bf, 1);

  // GraphConv layer 2
  zero_u32<<<2048, TPB, 0, stream>>>((unsigned int*)aggF, (long long)NN * DD);
  scatter_add<<<(EE * DD + TPB - 1) / TPB, TPB, 0, stream>>>(hF, src, dst, aggF);
  cvt_bf<<<(NN * DD + TPB - 1) / TPB, TPB, 0, stream>>>(aggF, agg_bf, NN * DD);
  gemm_wmma<<<NN / 16, 32, 0, stream>>>(h_bf, agg_bf, W2rT, W2lT, b2, nullptr, ne_bf, 0);

  // combined node+global Q/K/V biases
  gbias<<<1, 128, 0, stream>>>(g, WQg, bQg, bQ, cb);
  gbias<<<1, 128, 0, stream>>>(g, WKg, bKg, bK, cb + 128);
  gbias<<<1, 128, 0, stream>>>(g, WVg, bVg, bV, cb + 256);

  // Q/K/V projections (into padded bf16 buffers)
  gemm_wmma<<<NN / 16, 32, 0, stream>>>(ne_bf, nullptr, WQT, nullptr, cb,       nullptr, Qb, 0);
  gemm_wmma<<<NN / 16, 32, 0, stream>>>(ne_bf, nullptr, WKT, nullptr, cb + 128, nullptr, Kb, 0);
  gemm_wmma<<<NN / 16, 32, 0, stream>>>(ne_bf, nullptr, WVT, nullptr, cb + 256, nullptr, Vb, 0);
  vtrans<<<(NN * DD + TPB - 1) / TPB, TPB, 0, stream>>>(Vb, VTb);

  // flash attention + row-mean accumulation
  flash_attn<<<NN / 16, 32, 0, stream>>>(Qb, Kb, VTb, accum);

  // final MLP head + softmax
  head<<<1, 128, 0, stream>>>(accum, Wo, bo, Wfc1, bfc1, Wfc2, bfc2, Wfc3, bfc3, out);
}
